// SharedBackbone_72060961292569
// MI455X (gfx1250) — matrix operand order repaired and checked by reading, not compiled, not run on hardware
//
#include <hip/hip_runtime.h>
#include <hip/hip_fp16.h>
#include <math.h>

typedef __attribute__((ext_vector_type(16))) _Float16 v16h;
typedef __attribute__((ext_vector_type(8)))  _Float16 v8h;
typedef __attribute__((ext_vector_type(8)))  float    v8f;
typedef __attribute__((ext_vector_type(4)))  int      v4i;

typedef __attribute__((address_space(1))) v4i* gvec_p;   // global <4 x i32>*
typedef __attribute__((address_space(3))) v4i* lvec_p;   // LDS    <4 x i32>*

#define HEADS 4
#define DHEAD 32
#define HID 128
#define NGRAPH 16
#define NEG_SLOPE 0.2f
#define LN_EPS 1e-5f
#define MT 4                 // M-tiles (of 16 rows) per block -> 64 rows/block

// ---------------- utility kernels ----------------

__global__ void fill_f32(float* __restrict__ p, float v, long long n) {
    long long i = (long long)blockIdx.x * blockDim.x + threadIdx.x;
    if (i < n) p[i] = v;
}

__global__ void f32_to_f16(const float* __restrict__ in, _Float16* __restrict__ out, long long n) {
    long long i = (long long)blockIdx.x * blockDim.x + threadIdx.x;
    if (i < n) out[i] = (_Float16)in[i];
}

// W [K=128][N=128] row-major fp32 -> Wt [N=128][K=128] f16
__global__ void transpose_w(const float* __restrict__ W, _Float16* __restrict__ Wt) {
    int n = blockIdx.x * 16 + threadIdx.x;
    int k = blockIdx.y * 16 + threadIdx.y;
    Wt[(size_t)n * HID + k] = (_Float16)W[(size_t)k * HID + n];
}

// ---------------- WMMA GEMM: H[N,128] = X16[N,128] @ Wt16^T ----------------
// Block = 256 threads = 8 waves; wave w owns cols [16w,16w+16) for MT row-tiles.
// Both W^T (32 KB) and the 64-row A slab (16 KB) are staged in LDS with the
// CDNA5 async copy path (global_load_async_to_lds_b128 / ASYNCcnt); WMMA
// operands then come from LDS (ds_load_b128). B fragments are hoisted into
// registers once per wave and reused across the 4 M-tiles.
__global__ void gemm_wmma(const _Float16* __restrict__ X, const _Float16* __restrict__ Wt,
                          float* __restrict__ H, int Nn) {
    __shared__ __align__(32) _Float16 sW[HID * HID];      // 32 KB
    __shared__ __align__(32) _Float16 sA[MT * 16 * HID];  // 16 KB

    const int tid = threadIdx.x;
    const long long rowbase = (long long)blockIdx.x * (MT * 16);

#if __has_builtin(__builtin_amdgcn_global_load_async_to_lds_b128)
    {
        // stage W^T: 256 lanes x 16B x 8 iters = 32 KB
#pragma unroll
        for (int it = 0; it < (HID * HID) / (256 * 8); ++it) {
            __builtin_amdgcn_global_load_async_to_lds_b128(
                (gvec_p)(Wt + it * 2048 + tid * 8),
                (lvec_p)(sW + it * 2048 + tid * 8),
                0, 0);
        }
        // stage A slab: 64 rows x 128 K f16 = 16 KB = 256 lanes x 16B x 4 iters
#pragma unroll
        for (int it = 0; it < (MT * 16 * HID) / (256 * 8); ++it) {
            const int flat = it * 2048 + tid * 8;     // half index into sA
            const int lr = flat / HID;                // local row 0..63
            const int kk = flat % HID;
            long long gr = rowbase + lr;
            if (gr >= Nn) gr = Nn - 1;                // clamp (duplicates are fine)
            __builtin_amdgcn_global_load_async_to_lds_b128(
                (gvec_p)(X + gr * HID + kk),
                (lvec_p)(sA + flat),
                0, 0);
        }
#if __has_builtin(__builtin_amdgcn_s_wait_asynccnt)
        __builtin_amdgcn_s_wait_asynccnt(0);
#else
        asm volatile("s_wait_asynccnt 0x0" ::: "memory");
#endif
    }
#else
    for (int i = tid * 8; i < HID * HID; i += 256 * 8)
        *(v8h*)(sW + i) = *(const v8h*)(Wt + i);
    for (int f = tid * 8; f < MT * 16 * HID; f += 256 * 8) {
        long long gr = rowbase + f / HID;
        if (gr >= Nn) gr = Nn - 1;
        *(v8h*)(sA + f) = *(const v8h*)(X + gr * HID + (f % HID));
    }
#endif
    __syncthreads();

    const int wave = tid >> 5;
    const int lane = tid & 31;
    const int hi   = lane >> 4;       // lane half
    const int lo16 = lane & 15;
    const int col  = wave * 16 + lo16;

    // B fragments from LDS, loaded once and reused for all MT tiles.
    // B 32x16 f16 layout: lane holds column (lane&15), K = hi*16 .. hi*16+15
    v16h bfrag[4];
#pragma unroll
    for (int kt = 0; kt < 4; ++kt)
        bfrag[kt] = *(const v16h*)(sW + (size_t)col * HID + kt * 32 + hi * 16);

#pragma unroll
    for (int mt = 0; mt < MT; ++mt) {
        const int lrow = mt * 16 + lo16;
        v8f c = {};
#pragma unroll
        for (int kt = 0; kt < 4; ++kt) {
            const int k0 = kt * 32;
            // A 16x32 f16 layout: lane holds K[hi*16..hi*16+15] (matches B)
            v8h a0 = *(const v8h*)(sA + (size_t)lrow * HID + k0 + hi * 16);
            v8h a1 = *(const v8h*)(sA + (size_t)lrow * HID + k0 + hi * 16 + 8);
            v16h a;
#pragma unroll
            for (int i = 0; i < 8; ++i) { a[i] = a0[i]; a[8 + i] = a1[i]; }
            c = __builtin_amdgcn_wmma_f32_16x16x32_f16(false, a, false, bfrag[kt],
                                                       (short)0, c, false, false);
        }
        const long long ob = rowbase + mt * 16;
        if (ob + 16 <= Nn) {
            // full tile: unguarded clause-grouped stores
#pragma unroll
            for (int r = 0; r < 8; ++r)
                H[(size_t)(ob + r + 8 * hi) * HID + col] = c[r];
        } else {
#pragma unroll
            for (int r = 0; r < 8; ++r) {
                long long orow = ob + r + 8 * hi;    // C/D layout: M = r + 8*hi
                if (orow < Nn) H[(size_t)orow * HID + col] = c[r];
            }
        }
    }
}

// ---------------- per-node attention coefficients ----------------
// block = 128 (4 waves); wave w = head w; lane handles one of 32 dims.
__global__ void node_alphas(const float* __restrict__ H, const float* __restrict__ a_src,
                            const float* __restrict__ a_dst, float* __restrict__ as,
                            float* __restrict__ ad) {
    const int n = blockIdx.x;
    const int h = threadIdx.x >> 5;
    const int lane = threadIdx.x & 31;
    float hv = H[(size_t)n * HID + h * DHEAD + lane];
    float s = hv * a_src[h * DHEAD + lane];
    float d = hv * a_dst[h * DHEAD + lane];
#pragma unroll
    for (int off = 16; off > 0; off >>= 1) {
        s += __shfl_xor(s, off, 32);
        d += __shfl_xor(d, off, 32);
    }
    if (lane == 0) { as[n * HEADS + h] = s; ad[n * HEADS + h] = d; }
}

// ---------------- edge helpers ----------------
__device__ __forceinline__ void get_edge(const int* __restrict__ ei, long long E,
                                         long long e, int& s, int& d) {
    if (e < E) { s = ei[e]; d = ei[E + e]; }
    else       { s = d = (int)(e - E); }     // self-loop
}

__device__ __forceinline__ float atomic_max_f32(float* addr, float v) {
    if (v >= 0.0f) return __int_as_float(atomicMax((int*)addr, __float_as_int(v)));
    return __uint_as_float(atomicMin((unsigned int*)addr, __float_as_uint(v)));
}

// pass A: leaky-relu logit + scatter max  (thread per edge*head)
__global__ void edge_max(const int* __restrict__ ei, long long E, long long T4,
                         const float* __restrict__ as, const float* __restrict__ ad,
                         float* __restrict__ emax) {
    long long i = (long long)blockIdx.x * blockDim.x + threadIdx.x;
    if (i >= T4) return;
    long long e = i >> 2; int h = (int)(i & 3);
    int s, d; get_edge(ei, E, e, s, d);
    float v = as[(size_t)s * HEADS + h] + ad[(size_t)d * HEADS + h];
    v = v > 0.0f ? v : NEG_SLOPE * v;
    atomic_max_f32(&emax[(size_t)d * HEADS + h], v);
}

// pass B: exp(e - max) stored per edge + scatter sum
__global__ void edge_exp(const int* __restrict__ ei, long long E, long long T4,
                         const float* __restrict__ as, const float* __restrict__ ad,
                         const float* __restrict__ emax, float* __restrict__ exbuf,
                         float* __restrict__ esum) {
    long long i = (long long)blockIdx.x * blockDim.x + threadIdx.x;
    if (i >= T4) return;
    long long e = i >> 2; int h = (int)(i & 3);
    int s, d; get_edge(ei, E, e, s, d);
    float v = as[(size_t)s * HEADS + h] + ad[(size_t)d * HEADS + h];
    v = v > 0.0f ? v : NEG_SLOPE * v;
    float ex = __expf(v - emax[(size_t)d * HEADS + h]);
    exbuf[i] = ex;
    atomicAdd(&esum[(size_t)d * HEADS + h], ex);
}

// pass C: weighted aggregation. One wave per edge; lane carries 4 features (float4).
__global__ void edge_agg(const int* __restrict__ ei, long long E, long long TE,
                         const float* __restrict__ H, const float* __restrict__ exbuf,
                         const float* __restrict__ esum, float* __restrict__ agg) {
    long long e = (long long)blockIdx.x * 8 + (threadIdx.x >> 5);
    if (e >= TE) return;
    const int lane = threadIdx.x & 31;
    int s, d; get_edge(ei, E, e, s, d);
    const int head = lane >> 3;                      // 8 lanes (32 feats) per head
    float alpha = exbuf[e * HEADS + head] / esum[(size_t)d * HEADS + head];
    const float4 hv = *(const float4*)(H + (size_t)s * HID + lane * 4);
    float* ap = agg + (size_t)d * HID + lane * 4;
    atomicAdd(ap + 0, hv.x * alpha);
    atomicAdd(ap + 1, hv.y * alpha);
    atomicAdd(ap + 2, hv.z * alpha);
    atomicAdd(ap + 3, hv.w * alpha);
}

// ---------------- bias + LayerNorm + ELU (optionally emit f16 for next GEMM) ----------------
__global__ void ln_elu(const float* __restrict__ agg, const float* __restrict__ b,
                       const float* __restrict__ g, const float* __restrict__ be,
                       float* __restrict__ outf, int ostride, _Float16* __restrict__ outh) {
    __shared__ float sred[HID];
    const int n = blockIdx.x;
    const int t = threadIdx.x;
    float v = agg[(size_t)n * HID + t] + b[t];
    sred[t] = v; __syncthreads();
#pragma unroll
    for (int off = 64; off > 0; off >>= 1) { if (t < off) sred[t] += sred[t + off]; __syncthreads(); }
    float mu = sred[0] * (1.0f / HID); __syncthreads();
    float dv = v - mu;
    sred[t] = dv * dv; __syncthreads();
#pragma unroll
    for (int off = 64; off > 0; off >>= 1) { if (t < off) sred[t] += sred[t + off]; __syncthreads(); }
    float var = sred[0] * (1.0f / HID);
    float o = dv * rsqrtf(var + LN_EPS) * g[t] + be[t];
    o = o > 0.0f ? o : (__expf(o) - 1.0f);           // ELU
    if (outf) outf[(size_t)n * ostride + t] = o;
    if (outh) outh[(size_t)n * HID + t] = (_Float16)o;
}

// ---------------- global mean readout ----------------
__global__ void readout_accum(const float* __restrict__ out, const int* __restrict__ batch,
                              float* __restrict__ gsum, float* __restrict__ gcnt) {
    const int n = blockIdx.x;
    const int t = threadIdx.x;
    const int bg = batch[n];
    atomicAdd(&gsum[(size_t)bg * HID + t], out[(size_t)n * 256 + t]);
    if (t == 0) atomicAdd(&gcnt[bg], 1.0f);
}

__global__ void gdiv(float* __restrict__ gsum, const float* __restrict__ gcnt) {
    const int b = blockIdx.x;
    const int t = threadIdx.x;
    float c = gcnt[b]; c = c > 1.0f ? c : 1.0f;
    gsum[(size_t)b * HID + t] /= c;
}

__global__ void write_global(const float* __restrict__ gsum, const int* __restrict__ batch,
                             float* __restrict__ out) {
    const int n = blockIdx.x;
    const int t = threadIdx.x;
    out[(size_t)n * 256 + 128 + t] = gsum[(size_t)batch[n] * HID + t];
}

// ---------------- host launch ----------------
static inline size_t align256(size_t x) { return (x + 255) & ~(size_t)255; }

extern "C" void kernel_launch(void* const* d_in, const int* in_sizes, int n_in,
                              void* d_out, int out_size, void* d_ws, size_t ws_size,
                              hipStream_t stream) {
    const float* x      = (const float*)d_in[0];
    const int*   ei     = (const int*)d_in[1];     // [2,E] int32
    const int*   batch  = (const int*)d_in[2];
    const float* W1     = (const float*)d_in[3];
    const float* asrc1  = (const float*)d_in[4];
    const float* adst1  = (const float*)d_in[5];
    const float* b1     = (const float*)d_in[6];
    const float* g1     = (const float*)d_in[7];
    const float* be1    = (const float*)d_in[8];
    const float* W2     = (const float*)d_in[9];
    const float* asrc2  = (const float*)d_in[10];
    const float* adst2  = (const float*)d_in[11];
    const float* b2     = (const float*)d_in[12];
    const float* g2     = (const float*)d_in[13];
    const float* be2    = (const float*)d_in[14];
    float* out = (float*)d_out;

    const long long Nn = in_sizes[0] / HID;
    const long long E  = in_sizes[1] / 2;
    const long long TE = E + Nn;                   // edges incl. self-loops

    // workspace layout
    char* wsb = (char*)d_ws;
    size_t off = 0;
    _Float16* x16  = (_Float16*)(wsb + off); off = align256(off + (size_t)Nn * HID * 2);
    _Float16* wt16 = (_Float16*)(wsb + off); off = align256(off + (size_t)HID * HID * 2);
    float* hbuf  = (float*)(wsb + off); off = align256(off + (size_t)Nn * HID * 4);
    float* as    = (float*)(wsb + off); off = align256(off + (size_t)Nn * HEADS * 4);
    float* ad    = (float*)(wsb + off); off = align256(off + (size_t)Nn * HEADS * 4);
    float* emax  = (float*)(wsb + off); off = align256(off + (size_t)Nn * HEADS * 4);
    float* esum  = (float*)(wsb + off); off = align256(off + (size_t)Nn * HEADS * 4);
    float* exbuf = (float*)(wsb + off); off = align256(off + (size_t)TE * HEADS * 4);
    float* agg   = (float*)(wsb + off); off = align256(off + (size_t)Nn * HID * 4);
    float* gsum  = (float*)(wsb + off); off = align256(off + (size_t)NGRAPH * HID * 4);
    float* gcnt  = (float*)(wsb + off); off = align256(off + (size_t)NGRAPH * 4);
    (void)ws_size; (void)n_in; (void)out_size;

    const long long T4 = TE * HEADS;
    const int gblocks = (int)((Nn + (MT * 16) - 1) / (MT * 16));
    dim3 tblk(16, 16), tgrid(HID / 16, HID / 16);

    // ---- convert input to f16 ----
    f32_to_f16<<<(unsigned)((Nn * HID + 255) / 256), 256, 0, stream>>>(x, x16, Nn * HID);

    const float* Wl[2]  = { W1, W2 };
    const float* asl[2] = { asrc1, asrc2 };
    const float* adl[2] = { adst1, adst2 };
    const float* bl[2]  = { b1, b2 };
    const float* gl[2]  = { g1, g2 };
    const float* bel[2] = { be1, be2 };

    for (int layer = 0; layer < 2; ++layer) {
        transpose_w<<<tgrid, tblk, 0, stream>>>(Wl[layer], wt16);
        gemm_wmma<<<gblocks, 256, 0, stream>>>(x16, wt16, hbuf, (int)Nn);
        node_alphas<<<(unsigned)Nn, 128, 0, stream>>>(hbuf, asl[layer], adl[layer], as, ad);

        fill_f32<<<(unsigned)((Nn * HEADS + 255) / 256), 256, 0, stream>>>(emax, -INFINITY, Nn * HEADS);
        fill_f32<<<(unsigned)((Nn * HEADS + 255) / 256), 256, 0, stream>>>(esum, 0.0f, Nn * HEADS);
        fill_f32<<<(unsigned)((Nn * HID + 255) / 256), 256, 0, stream>>>(agg, 0.0f, Nn * HID);

        edge_max<<<(unsigned)((T4 + 255) / 256), 256, 0, stream>>>(ei, E, T4, as, ad, emax);
        edge_exp<<<(unsigned)((T4 + 255) / 256), 256, 0, stream>>>(ei, E, T4, as, ad, emax, exbuf, esum);
        edge_agg<<<(unsigned)((TE + 7) / 8), 256, 0, stream>>>(ei, E, TE, hbuf, exbuf, esum, agg);

        if (layer == 0) {
            // emit f16 activations for layer-2 WMMA GEMM (overwrites consumed x16)
            ln_elu<<<(unsigned)Nn, 128, 0, stream>>>(agg, bl[layer], gl[layer], bel[layer],
                                                     (float*)nullptr, 0, x16);
        } else {
            // final node features straight into d_out[:, 0:128]
            ln_elu<<<(unsigned)Nn, 128, 0, stream>>>(agg, bl[layer], gl[layer], bel[layer],
                                                     out, 256, (_Float16*)nullptr);
        }
    }

    // ---- per-graph mean readout -> d_out[:, 128:256] ----
    fill_f32<<<1, 256, 0, stream>>>(gsum, 0.0f, (long long)NGRAPH * HID);
    fill_f32<<<1, 32, 0, stream>>>(gcnt, 0.0f, NGRAPH);
    readout_accum<<<(unsigned)Nn, 128, 0, stream>>>(out, batch, gsum, gcnt);
    gdiv<<<NGRAPH, 128, 0, stream>>>(gsum, gcnt);
    write_global<<<(unsigned)Nn, 128, 0, stream>>>(gsum, batch, out);
}
